// Attention_local_31164282700682
// MI455X (gfx1250) — compile-verified, matrix operand order checked by reading, not run on hardware
//
#include <hip/hip_runtime.h>
#include <hip/hip_fp16.h>

typedef __attribute__((ext_vector_type(16))) _Float16 v16h;
typedef __attribute__((ext_vector_type(8)))  float    v8f;

#define HH    256
#define WWID  256
#define WIN   16
#define NGRID 121
#define NBOX  (121*121)
#define KEEP  50
#define DIM   128
#define HEADS 8
#define DH    64
#define INNER 512
#define QKVN  1536

// ---------------------------------------------------------------------------
// WMMA helpers (CDNA5 gfx1250, wave32, 16x16x32 f16 -> f32)
// Fragment layouts per CDNA5 ISA 7.12.2:
//   A/B frag elem i of lane L: row/col = L&15, k = (L>>4)*8 + (i<8 ? i : i+8)
//   C/D frag elem g of lane L: m = g + (L<16 ? 0 : 8), n = L&15
// ---------------------------------------------------------------------------
__device__ __forceinline__ v8f wmma16(v16h a, v16h b, v8f c) {
    return __builtin_amdgcn_wmma_f32_16x16x32_f16(false, a, false, b, (short)0, c,
                                                  false, false);
}

// Generic strided fragment loader: element i -> p[k*sk + (lane&15)*smn]
__device__ __forceinline__ v16h load_frag(const _Float16* __restrict__ p,
                                          int sk, int smn) {
    const int lane = threadIdx.x & 31;
    const int mn   = lane & 15;
    const int kb   = (lane >> 4) << 3;
    v16h f;
#pragma unroll
    for (int i = 0; i < 16; ++i) {
        const int k = kb + (i < 8 ? i : i + 8);
        f[i] = p[k * sk + mn * smn];
    }
    return f;
}

// ---------------------------------------------------------------------------
// CDNA5 async global->LDS copy (bypasses VGPRs, tracked on ASYNCcnt).
// GV mode: per-lane 64-bit global address, per-lane 32-bit LDS byte address.
// LDS aperture uses addr[31:0] as the LDS offset, so truncating the generic
// pointer of a __shared__ object yields the required VDST value.
// ---------------------------------------------------------------------------
__device__ __forceinline__ void async_ld_b128(void* lds_dst, const void* gsrc) {
    const unsigned lds = (unsigned)(uintptr_t)lds_dst;
    asm volatile("global_load_async_to_lds_b128 %0, %1, off"
                 :: "v"(lds), "v"(gsrc)
                 : "memory");
}
__device__ __forceinline__ void wait_async0() {
    asm volatile("s_wait_asynccnt 0x0" ::: "memory");
}

// ---------------------------------------------------------------------------
// 1) weight conversion f32 -> f16
// ---------------------------------------------------------------------------
__global__ void wcvt_kernel(const float* __restrict__ wqkv,
                            const float* __restrict__ wout,
                            _Float16* __restrict__ wqkvh,
                            _Float16* __restrict__ wouth) {
    const int i = blockIdx.x * blockDim.x + threadIdx.x;
    if (i < QKVN * DIM)  wqkvh[i] = (_Float16)wqkv[i];
    if (i < DIM * INNER) wouth[i] = (_Float16)wout[i];
}

// ---------------------------------------------------------------------------
// 2) entropy over 2-channel prob map (b,2,128,128)
// ---------------------------------------------------------------------------
__global__ void entropy_kernel(const float* __restrict__ prob,
                               float* __restrict__ ent, int total) {
    const int i = blockIdx.x * blockDim.x + threadIdx.x;
    if (i >= total) return;
    const int bi = i >> 14;           // /16384
    const int r  = i & 16383;
    const float p0 = prob[(size_t)bi * 32768 + r];
    const float p1 = prob[(size_t)bi * 32768 + 16384 + r];
    ent[i] = -(p0 * __log2f(p0 + 1e-10f) + p1 * __log2f(p1 + 1e-10f));
}

// ---------------------------------------------------------------------------
// 3) 8x8 VALID box filter with fix_w -> window scores (b, 121*121)
// ---------------------------------------------------------------------------
__global__ void score_kernel(const float* __restrict__ ent,
                             const float* __restrict__ fixw,
                             float* __restrict__ score, int total) {
    const int i = blockIdx.x * blockDim.x + threadIdx.x;
    if (i >= total) return;
    const int bi = i / NBOX;
    const int t  = i - bi * NBOX;
    const int p  = t / NGRID, q = t - p * NGRID;
    const float* e = ent + (size_t)bi * 16384 + p * 128 + q;
    float s = 0.f;
#pragma unroll
    for (int r = 0; r < 8; ++r)
#pragma unroll
        for (int c = 0; c < 8; ++c)
            s += e[r * 128 + c] * fixw[r * 8 + c];
    score[i] = s * (1.0f / 64.0f);
}

// ---------------------------------------------------------------------------
// 4) greedy NMS, one block per batch, matches jnp argmax (first max wins)
// ---------------------------------------------------------------------------
__global__ __launch_bounds__(256) void nms_kernel(const float* __restrict__ score,
                                                  int* __restrict__ sx,
                                                  int* __restrict__ sy) {
    const int bi = blockIdx.x, tid = threadIdx.x;
    __shared__ float sv[256];
    __shared__ int   si[256];
    __shared__ unsigned char live[NBOX];
    for (int i = tid; i < NBOX; i += 256) live[i] = 1;
    const float* s = score + (size_t)bi * NBOX;
    __syncthreads();
    for (int k = 0; k < KEEP; ++k) {
        float best = -3.4e38f; int bidx = 1 << 30;
        for (int i = tid; i < NBOX; i += 256) {
            if (live[i]) {
                const float v = s[i];
                if (v > best || (v == best && i < bidx)) { best = v; bidx = i; }
            }
        }
        sv[tid] = best; si[tid] = bidx;
        __syncthreads();
        for (int off = 128; off > 0; off >>= 1) {
            if (tid < off) {
                const float v2 = sv[tid + off]; const int i2 = si[tid + off];
                if (v2 > sv[tid] || (v2 == sv[tid] && i2 < si[tid])) {
                    sv[tid] = v2; si[tid] = i2;
                }
            }
            __syncthreads();
        }
        const int sel = si[0];
        if (tid == 0) {
            sx[bi * KEEP + k] = (sel % NGRID) * 2;
            sy[bi * KEEP + k] = (sel / NGRID) * 2;
        }
        const float bx1 = (float)((sel % NGRID) * 2), by1 = (float)((sel / NGRID) * 2);
        const float bx2 = bx1 + 15.f, by2 = by1 + 15.f;
        for (int i = tid; i < NBOX; i += 256) {
            if (!live[i]) continue;
            const float x1 = (float)((i % NGRID) * 2), y1 = (float)((i / NGRID) * 2);
            float iw = fminf(x1 + 15.f, bx2) - fmaxf(x1, bx1);
            float ih = fminf(y1 + 15.f, by2) - fmaxf(y1, by1);
            iw = fmaxf(iw, 0.f); ih = fmaxf(ih, 0.f);
            const float inter = iw * ih;                  // areas are 225 each
            if (inter / (450.f - inter) > 0.2f) live[i] = 0;
        }
        __syncthreads();
    }
}

// ---------------------------------------------------------------------------
// 5) bilinear crop -> f16 tokens: one block per kept window, thread = token
// ---------------------------------------------------------------------------
__global__ __launch_bounds__(256) void gather_kernel(const float* __restrict__ x,
                                                     const int* __restrict__ sx,
                                                     const int* __restrict__ sy,
                                                     _Float16* __restrict__ toks) {
    const int wi = blockIdx.x;
    const int bi = wi / KEEP;
    const int t  = threadIdx.x;
    const int i  = t >> 4, j = t & 15;
    const float oy = (i + 0.5f) * (15.0f / 16.0f);
    const float ox = (j + 0.5f) * (15.0f / 16.0f);
    const int   ly = (int)floorf(oy), lx = (int)floorf(ox);
    const float fy = oy - (float)ly,  fx = ox - (float)lx;
    const int r0 = sy[wi] + ly, c0 = sx[wi] + lx;
    const float* p00 = x + (((size_t)bi * 65536) + (size_t)(r0 * 256 + c0)) * DIM;
    const float* p01 = p00 + DIM;
    const float* p10 = p00 + 256 * DIM;
    const float* p11 = p10 + DIM;
    const float w00 = (1.f - fy) * (1.f - fx), w01 = (1.f - fy) * fx;
    const float w10 = fy * (1.f - fx),         w11 = fy * fx;
    _Float16* o = toks + ((size_t)wi * 256 + t) * DIM;
#pragma unroll 4
    for (int d = 0; d < DIM; d += 4) {
        const float4 a = *(const float4*)(p00 + d);
        const float4 b = *(const float4*)(p01 + d);
        const float4 c = *(const float4*)(p10 + d);
        const float4 e = *(const float4*)(p11 + d);
        o[d + 0] = (_Float16)(w00 * a.x + w01 * b.x + w10 * c.x + w11 * e.x);
        o[d + 1] = (_Float16)(w00 * a.y + w01 * b.y + w10 * c.y + w11 * e.y);
        o[d + 2] = (_Float16)(w00 * a.z + w01 * b.z + w10 * c.z + w11 * e.z);
        o[d + 3] = (_Float16)(w00 * a.w + w01 * b.w + w10 * c.w + w11 * e.w);
    }
}

// ---------------------------------------------------------------------------
// 6) WMMA GEMM: C[M,N] = A[M,K] @ W[N,K]^T  (both f16, f32 accumulate)
// Block tile 128x128, K-step 32, 8 waves (2x4), wave tile 64x32.
// Tiles staged with GLOBAL_LOAD_ASYNC_TO_LDS_B128 (ASYNCcnt), no VGPR bounce.
// MODE 0: store f16.  MODE 1: +bias, atomic scatter-add into output image.
// ---------------------------------------------------------------------------
template <int MODE>
__global__ __launch_bounds__(256) void gemm_wmma(
        const _Float16* __restrict__ A, const _Float16* __restrict__ W,
        _Float16* __restrict__ Ch,
        const float* __restrict__ bias, float* __restrict__ outAcc,
        const int* __restrict__ sx, const int* __restrict__ sy,
        int M, int N, int K) {
    __shared__ _Float16 As[128 * 32];
    __shared__ _Float16 Ws[128 * 32];
    const int tid  = threadIdx.x;
    const int lane = tid & 31;
    const int wave = tid >> 5;
    const int wm = wave & 1, wn = wave >> 1;
    const int m0 = blockIdx.y * 128, n0 = blockIdx.x * 128;

    const v8f vz = {0.f, 0.f, 0.f, 0.f, 0.f, 0.f, 0.f, 0.f};
    v8f acc[4][2];
#pragma unroll
    for (int i = 0; i < 4; ++i)
#pragma unroll
        for (int j = 0; j < 2; ++j) acc[i][j] = vz;

    for (int k0 = 0; k0 < K; k0 += 32) {
#pragma unroll
        for (int e = 0; e < 2; ++e) {              // 512 b128 copies per tile
            const int l = tid + e * 256;
            const int row = l >> 2, c = l & 3;
            async_ld_b128(((uint4*)As) + l,
                          (const uint4*)(A + (size_t)(m0 + row) * K + k0) + c);
            async_ld_b128(((uint4*)Ws) + l,
                          (const uint4*)(W + (size_t)(n0 + row) * K + k0) + c);
        }
        if (k0 + 32 < K)
            __builtin_prefetch(A + (size_t)(m0 + (tid >> 1)) * K + k0 + 32, 0, 1);
        wait_async0();
        __syncthreads();
        v16h af[4], wf[2];
#pragma unroll
        for (int mf = 0; mf < 4; ++mf)
            af[mf] = load_frag(As + (wm * 64 + mf * 16) * 32, 1, 32);
#pragma unroll
        for (int nf = 0; nf < 2; ++nf)
            wf[nf] = load_frag(Ws + (wn * 32 + nf * 16) * 32, 1, 32);
#pragma unroll
        for (int mf = 0; mf < 4; ++mf)
#pragma unroll
            for (int nf = 0; nf < 2; ++nf)
                acc[mf][nf] = wmma16(af[mf], wf[nf], acc[mf][nf]);
        __syncthreads();
    }

#pragma unroll
    for (int mf = 0; mf < 4; ++mf)
#pragma unroll
        for (int nf = 0; nf < 2; ++nf)
#pragma unroll
            for (int g = 0; g < 8; ++g) {
                const int row = m0 + wm * 64 + mf * 16 + ((lane < 16) ? g : g + 8);
                const int col = n0 + wn * 32 + nf * 16 + (lane & 15);
                const float v = acc[mf][nf][g];
                if (MODE == 0) {
                    Ch[(size_t)row * N + col] = (_Float16)v;
                } else {
                    const int wi = row >> 8;           // window index
                    const int t  = row & 255;
                    const int y  = sy[wi] + (t >> 4);
                    const int xc = sx[wi] + (t & 15);
                    const int bi = wi / KEEP;
                    atomicAdd(outAcc + (((size_t)bi * 65536 +
                                         (size_t)(y * 256 + xc)) * DIM + col),
                              v + bias[col]);
                }
            }
}

// ---------------------------------------------------------------------------
// 7) attention per (head, window). 8 waves; wave owns 32 query rows.
// Two passes: (a) online row max/sum over S = q k^T / 8, (b) O = P V.
// K tile staged with async-to-LDS; S comes out of WMMA C-frags; rows reduced
// with half-wave shfl_xor; P bounced through per-wave LDS (C->A layout).
// ---------------------------------------------------------------------------
__global__ __launch_bounds__(256) void attn_kernel(const _Float16* __restrict__ qkv,
                                                   _Float16* __restrict__ aout) {
    const int h = blockIdx.x;
    const int m = blockIdx.y;
    const int tid = threadIdx.x, lane = tid & 31, w = tid >> 5;
    __shared__ _Float16 kld[256 * DH];        // K matrix for this head (32 KB)
    __shared__ _Float16 Ps[8][32 * 32];       // per-wave P scratch   (16 KB)

    {   // cooperative K load: thread r copies its 64-half row (8x async b128)
        const uint4* src = (const uint4*)(qkv + ((size_t)m * 256 + tid) * QKVN +
                                          INNER + h * DH);
        uint4* dst = (uint4*)(kld + tid * DH);
#pragma unroll
        for (int c = 0; c < 8; ++c) async_ld_b128(dst + c, src + c);
    }
    wait_async0();
    __syncthreads();

    const _Float16* qbase = qkv + ((size_t)m * 256 + w * 32) * QKVN + h * DH;
    const _Float16* vbase = qkv + ((size_t)m * 256) * QKVN + 2 * INNER + h * DH;
    const float scale = 0.125f;               // 1/sqrt(64)
    const v8f vz = {0.f, 0.f, 0.f, 0.f, 0.f, 0.f, 0.f, 0.f};

    float rmax[2][8], rsum[2][8];
#pragma unroll
    for (int mt = 0; mt < 2; ++mt)
#pragma unroll
        for (int g = 0; g < 8; ++g) { rmax[mt][g] = -1e30f; rsum[mt][g] = 0.f; }

    // ---- pass 1: row statistics -------------------------------------------
    for (int n0 = 0; n0 < 256; n0 += 32) {
        v8f s[2][2] = {{vz, vz}, {vz, vz}};
#pragma unroll
        for (int kk = 0; kk < DH; kk += 32) {
            const v16h a0 = load_frag(qbase + kk, 1, QKVN);
            const v16h a1 = load_frag(qbase + 16 * QKVN + kk, 1, QKVN);
            const v16h b0 = load_frag(kld + n0 * DH + kk, 1, DH);
            const v16h b1 = load_frag(kld + (n0 + 16) * DH + kk, 1, DH);
            s[0][0] = wmma16(a0, b0, s[0][0]);
            s[0][1] = wmma16(a0, b1, s[0][1]);
            s[1][0] = wmma16(a1, b0, s[1][0]);
            s[1][1] = wmma16(a1, b1, s[1][1]);
        }
#pragma unroll
        for (int mt = 0; mt < 2; ++mt)
#pragma unroll
            for (int g = 0; g < 8; ++g) {
                const float v0 = s[mt][0][g] * scale;
                const float v1 = s[mt][1][g] * scale;
                float cm = fmaxf(v0, v1);
#pragma unroll
                for (int o = 1; o < 16; o <<= 1) cm = fmaxf(cm, __shfl_xor(cm, o, 32));
                const float nm = fmaxf(rmax[mt][g], cm);
                float es = __expf(v0 - nm) + __expf(v1 - nm);
#pragma unroll
                for (int o = 1; o < 16; o <<= 1) es += __shfl_xor(es, o, 32);
                rsum[mt][g] = rsum[mt][g] * __expf(rmax[mt][g] - nm) + es;
                rmax[mt][g] = nm;
            }
    }

    float inv[2][8];
#pragma unroll
    for (int mt = 0; mt < 2; ++mt)
#pragma unroll
        for (int g = 0; g < 8; ++g) inv[mt][g] = 1.0f / rsum[mt][g];

    // ---- pass 2: O = softmax(S) @ V ---------------------------------------
    v8f oacc[2][4];
#pragma unroll
    for (int mt = 0; mt < 2; ++mt)
#pragma unroll
        for (int dt = 0; dt < 4; ++dt) oacc[mt][dt] = vz;

    for (int n0 = 0; n0 < 256; n0 += 32) {
        v8f s[2][2] = {{vz, vz}, {vz, vz}};
#pragma unroll
        for (int kk = 0; kk < DH; kk += 32) {
            const v16h a0 = load_frag(qbase + kk, 1, QKVN);
            const v16h a1 = load_frag(qbase + 16 * QKVN + kk, 1, QKVN);
            const v16h b0 = load_frag(kld + n0 * DH + kk, 1, DH);
            const v16h b1 = load_frag(kld + (n0 + 16) * DH + kk, 1, DH);
            s[0][0] = wmma16(a0, b0, s[0][0]);
            s[0][1] = wmma16(a0, b1, s[0][1]);
            s[1][0] = wmma16(a1, b0, s[1][0]);
            s[1][1] = wmma16(a1, b1, s[1][1]);
        }
        const int rlo = (lane < 16) ? 0 : 8;
#pragma unroll
        for (int mt = 0; mt < 2; ++mt)
#pragma unroll
            for (int nt = 0; nt < 2; ++nt)
#pragma unroll
                for (int g = 0; g < 8; ++g) {
                    const float p = __expf(s[mt][nt][g] * scale - rmax[mt][g]) *
                                    inv[mt][g];
                    Ps[w][(mt * 16 + g + rlo) * 32 + nt * 16 + (lane & 15)] =
                        (_Float16)p;
                }
        __syncthreads();
        const v16h pa0 = load_frag(&Ps[w][0],       1, 32);
        const v16h pa1 = load_frag(&Ps[w][16 * 32], 1, 32);
#pragma unroll
        for (int dt = 0; dt < 4; ++dt) {
            // V B-frag: k rows run down the token dim (stride QKVN), n = d col
            const v16h bv = load_frag(vbase + (size_t)n0 * QKVN + dt * 16, QKVN, 1);
            oacc[0][dt] = wmma16(pa0, bv, oacc[0][dt]);
            oacc[1][dt] = wmma16(pa1, bv, oacc[1][dt]);
        }
        __syncthreads();
    }

#pragma unroll
    for (int mt = 0; mt < 2; ++mt)
#pragma unroll
        for (int dt = 0; dt < 4; ++dt)
#pragma unroll
            for (int g = 0; g < 8; ++g) {
                const int row = m * 256 + w * 32 + mt * 16 + ((lane < 16) ? g : g + 8);
                const int col = h * DH + dt * 16 + (lane & 15);
                aout[(size_t)row * INNER + col] = (_Float16)oacc[mt][dt][g];
            }
}

// ---------------------------------------------------------------------------
// 8) coverage counts + final residual combine
// ---------------------------------------------------------------------------
__global__ __launch_bounds__(256) void count_kernel(const int* __restrict__ sx,
                                                    const int* __restrict__ sy,
                                                    float* __restrict__ cnt) {
    const int wi = blockIdx.x, bi = wi / KEEP, t = threadIdx.x;
    const int y = sy[wi] + (t >> 4), x = sx[wi] + (t & 15);
    atomicAdd(cnt + (size_t)bi * 65536 + y * 256 + x, 1.0f);
}

__global__ void finalize_kernel(const float* __restrict__ x,
                                const float* __restrict__ cnt,
                                float* __restrict__ out, int total) {
    const int e = blockIdx.x * blockDim.x + threadIdx.x;
    if (e >= total) return;
    const int pix = e >> 7;   // /DIM
    out[e] = x[e] + out[e] / (cnt[pix] + 1e-10f);
}

// ---------------------------------------------------------------------------
// host launcher
// ---------------------------------------------------------------------------
extern "C" void kernel_launch(void* const* d_in, const int* in_sizes, int n_in,
                              void* d_out, int out_size, void* d_ws, size_t ws_size,
                              hipStream_t stream) {
    const float* x     = (const float*)d_in[0];
    const float* prob  = (const float*)d_in[1];
    const float* fixw  = (const float*)d_in[2];
    const float* wqkv  = (const float*)d_in[3];
    const float* wout  = (const float*)d_in[4];
    const float* bout  = (const float*)d_in[5];
    float* out = (float*)d_out;

    const int b = in_sizes[0] / (HH * WWID * DIM);
    const int M = b * KEEP * 256;               // token rows

    // carve workspace (256B aligned regions)
    char* p = (char*)d_ws;
    auto take = [&](size_t bytes) -> char* {
        char* r = p;
        p += (bytes + 255) & ~(size_t)255;
        return r;
    };
    float*    ent    = (float*)take((size_t)b * 16384 * 4);
    float*    score  = (float*)take((size_t)b * NBOX * 4);
    int*      sx     = (int*)take((size_t)b * KEEP * 4);
    int*      sy     = (int*)take((size_t)b * KEEP * 4);
    _Float16* wqkvh  = (_Float16*)take((size_t)QKVN * DIM * 2);
    _Float16* wouth  = (_Float16*)take((size_t)DIM * INNER * 2);
    _Float16* toks   = (_Float16*)take((size_t)M * DIM * 2);
    _Float16* qkvb   = (_Float16*)take((size_t)M * QKVN * 2);
    _Float16* aoutb  = (_Float16*)take((size_t)M * INNER * 2);
    float*    cnt    = (float*)take((size_t)b * 65536 * 4);

    hipMemsetAsync(d_out, 0, (size_t)out_size * 4, stream);
    hipMemsetAsync(cnt, 0, (size_t)b * 65536 * 4, stream);

    wcvt_kernel<<<(QKVN * DIM + 255) / 256, 256, 0, stream>>>(wqkv, wout, wqkvh, wouth);

    const int entN = b * 16384;
    entropy_kernel<<<(entN + 255) / 256, 256, 0, stream>>>(prob, ent, entN);

    const int scN = b * NBOX;
    score_kernel<<<(scN + 255) / 256, 256, 0, stream>>>(ent, fixw, score, scN);

    nms_kernel<<<b, 256, 0, stream>>>(score, sx, sy);

    gather_kernel<<<b * KEEP, 256, 0, stream>>>(x, sx, sy, toks);

    // QKV projection: [M,128] @ [1536,128]^T -> f16 [M,1536]
    gemm_wmma<0><<<dim3(QKVN / 128, M / 128), 256, 0, stream>>>(
        toks, wqkvh, qkvb, nullptr, nullptr, nullptr, nullptr, M, QKVN, DIM);

    attn_kernel<<<dim3(HEADS, b * KEEP), 256, 0, stream>>>(qkvb, aoutb);

    // output projection + bias + fused scatter-add into image accumulator
    gemm_wmma<1><<<dim3(DIM / 128, M / 128), 256, 0, stream>>>(
        aoutb, wouth, nullptr, bout, out, sx, sy, M, DIM, INNER);

    count_kernel<<<b * KEEP, 256, 0, stream>>>(sx, sy, cnt);

    finalize_kernel<<<(out_size + 255) / 256, 256, 0, stream>>>(x, cnt, out, out_size);
}
